// ValueGRU_17360257811169
// MI455X (gfx1250) — matrix operand order repaired and checked by reading, not compiled, not run on hardware
//
#include <hip/hip_runtime.h>

#define B_   64
#define T_   1024
#define OBS_ 128
#define H_   512
#define H3_  1536
#define M_   (B_ * T_)     // 65536 rows of (b,t)
#define NWG_GRU 32

typedef __attribute__((ext_vector_type(16))) __bf16 v16bf;
typedef __attribute__((ext_vector_type(8)))  __bf16 bf16x8;
typedef __attribute__((ext_vector_type(8)))  float  v8f;

// ---------- helpers ----------------------------------------------------------

static __device__ __forceinline__ unsigned short f2bf(float f) {  // RNE f32->bf16
  unsigned int u = __float_as_uint(f);
  u += 0x7fffu + ((u >> 16) & 1u);
  return (unsigned short)(u >> 16);
}

static __device__ __forceinline__ bf16x8 ld8(const unsigned short* p) {
  return *reinterpret_cast<const bf16x8*>(p);   // 16B -> global_load_b128 / ds_load_b128
}

static __device__ __forceinline__ v16bf cat8(bf16x8 lo, bf16x8 hi) {
  v16bf r;
#pragma unroll
  for (int i = 0; i < 8; ++i) { r[i] = lo[i]; r[i + 8] = hi[i]; }
  return r;
}

// A tile 16(M)x32(K) from row-major X (ld = ldk elems).
// Layout: lanes 0-15 hold K 0..15; lanes 16-31 hold K 16..31 (matches B).
static __device__ __forceinline__ v16bf load_a16x32(const unsigned short* X, int m0, int ldk,
                                                    int kbase, int lrow, int ha) {
  const unsigned short* p = X + (size_t)(m0 + lrow) * ldk + kbase + ha * 16;
  return cat8(ld8(p), ld8(p + 8));
}

// B tile 32(K)x16(N) where column n of B is row (n0+n) of row-major weight W (N x K).
// Layout (contiguous k, same as A): lanes 0-15 -> K 0..15 of col lane; lanes 16-31 -> K 16..31.
static __device__ __forceinline__ v16bf load_b32x16(const unsigned short* W, int n0, int ldk,
                                                    int kbase, int lrow, int ha) {
  const unsigned short* p = W + (size_t)(n0 + lrow) * ldk + kbase + ha * 16;
  return cat8(ld8(p), ld8(p + 8));
}

#define WMMA_BF16(a, b, c) \
  __builtin_amdgcn_wmma_f32_16x16x32_bf16(false, (a), false, (b), (short)0, (c), false, false)

// ---------- small utility kernels -------------------------------------------

__global__ void k_cvt(const float* __restrict__ s, unsigned short* __restrict__ d, int n) {
  int i = blockIdx.x * 256 + threadIdx.x;
  if (i < n) d[i] = f2bf(s[i]);
}

__global__ void k_init(float* __restrict__ out, const float* __restrict__ b_val,
                       unsigned int* __restrict__ hz, unsigned int* __restrict__ counter) {
  int i = blockIdx.x * 256 + threadIdx.x;
  if (i < M_) out[i] = b_val[0];                 // value head bias, atomics accumulate on top
  if (i < (2 * B_ * H_ / 2)) hz[i] = 0u;         // zero both bf16 h buffers (h0 = 0)
  if (i == 0) *counter = 0u;                     // grid barrier counter
}

// ---------- x = ReLU(obs @ W_emb^T + b_emb), bf16 out -----------------------
// one wave computes a 16x64 output strip (4 N-tiles), K = 128, software-pipelined

__global__ void __launch_bounds__(128) k_embed(const unsigned short* __restrict__ obs_bf,
                                               const unsigned short* __restrict__ Wemb_bf,
                                               const float* __restrict__ b_emb,
                                               unsigned short* __restrict__ x_bf) {
  int wid  = blockIdx.x * 4 + (threadIdx.x >> 5);
  int lane = threadIdx.x & 31, lrow = lane & 15, ha = lane >> 4;
  int m0 = (wid >> 3) * 16;
  int n0 = (wid & 7) * 64;
  v8f acc[4] = {};
  v16bf a  = load_a16x32(obs_bf, m0, OBS_, 0, lrow, ha);
  v16bf b0 = load_b32x16(Wemb_bf, n0 + 0,  OBS_, 0, lrow, ha);
  v16bf b1 = load_b32x16(Wemb_bf, n0 + 16, OBS_, 0, lrow, ha);
  v16bf b2 = load_b32x16(Wemb_bf, n0 + 32, OBS_, 0, lrow, ha);
  v16bf b3 = load_b32x16(Wemb_bf, n0 + 48, OBS_, 0, lrow, ha);
#pragma unroll
  for (int kk = 0; kk < 3; ++kk) {               // pipelined: fetch k+1 before k's WMMAs
    int kb = (kk + 1) * 32;
    v16bf a2 = load_a16x32(obs_bf, m0, OBS_, kb, lrow, ha);
    v16bf c0 = load_b32x16(Wemb_bf, n0 + 0,  OBS_, kb, lrow, ha);
    v16bf c1 = load_b32x16(Wemb_bf, n0 + 16, OBS_, kb, lrow, ha);
    v16bf c2 = load_b32x16(Wemb_bf, n0 + 32, OBS_, kb, lrow, ha);
    v16bf c3 = load_b32x16(Wemb_bf, n0 + 48, OBS_, kb, lrow, ha);
    acc[0] = WMMA_BF16(a, b0, acc[0]);
    acc[1] = WMMA_BF16(a, b1, acc[1]);
    acc[2] = WMMA_BF16(a, b2, acc[2]);
    acc[3] = WMMA_BF16(a, b3, acc[3]);
    a = a2; b0 = c0; b1 = c1; b2 = c2; b3 = c3;
  }
  acc[0] = WMMA_BF16(a, b0, acc[0]);
  acc[1] = WMMA_BF16(a, b1, acc[1]);
  acc[2] = WMMA_BF16(a, b2, acc[2]);
  acc[3] = WMMA_BF16(a, b3, acc[3]);
#pragma unroll
  for (int j = 0; j < 4; ++j) {
    int n = n0 + j * 16 + lrow;
    float bias = b_emb[n];
#pragma unroll
    for (int r = 0; r < 8; ++r) {
      int m = m0 + r + ha * 8;
      float v = acc[j][r] + bias;
      x_bf[(size_t)m * H_ + n] = f2bf(v > 0.f ? v : 0.f);
    }
  }
}

// ---------- gx = x @ W_ih^T + b_ih, f32 out in (T, B, 3H) layout ------------

__global__ void __launch_bounds__(128) k_gx(const unsigned short* __restrict__ x_bf,
                                            const unsigned short* __restrict__ Wih_bf,
                                            const float* __restrict__ b_ih,
                                            float* __restrict__ gx) {
  int wid  = blockIdx.x * 4 + (threadIdx.x >> 5);
  int lane = threadIdx.x & 31, lrow = lane & 15, ha = lane >> 4;
  int m0 = (wid / 24) * 16;
  int n0 = (wid % 24) * 64;
  v8f acc[4] = {};
  v16bf a  = load_a16x32(x_bf, m0, H_, 0, lrow, ha);
  v16bf b0 = load_b32x16(Wih_bf, n0 + 0,  H_, 0, lrow, ha);
  v16bf b1 = load_b32x16(Wih_bf, n0 + 16, H_, 0, lrow, ha);
  v16bf b2 = load_b32x16(Wih_bf, n0 + 32, H_, 0, lrow, ha);
  v16bf b3 = load_b32x16(Wih_bf, n0 + 48, H_, 0, lrow, ha);
#pragma unroll
  for (int kk = 0; kk < 15; ++kk) {              // pipelined K loop
    int kb = (kk + 1) * 32;
    v16bf a2 = load_a16x32(x_bf, m0, H_, kb, lrow, ha);
    v16bf c0 = load_b32x16(Wih_bf, n0 + 0,  H_, kb, lrow, ha);
    v16bf c1 = load_b32x16(Wih_bf, n0 + 16, H_, kb, lrow, ha);
    v16bf c2 = load_b32x16(Wih_bf, n0 + 32, H_, kb, lrow, ha);
    v16bf c3 = load_b32x16(Wih_bf, n0 + 48, H_, kb, lrow, ha);
    acc[0] = WMMA_BF16(a, b0, acc[0]);
    acc[1] = WMMA_BF16(a, b1, acc[1]);
    acc[2] = WMMA_BF16(a, b2, acc[2]);
    acc[3] = WMMA_BF16(a, b3, acc[3]);
    a = a2; b0 = c0; b1 = c1; b2 = c2; b3 = c3;
  }
  acc[0] = WMMA_BF16(a, b0, acc[0]);
  acc[1] = WMMA_BF16(a, b1, acc[1]);
  acc[2] = WMMA_BF16(a, b2, acc[2]);
  acc[3] = WMMA_BF16(a, b3, acc[3]);
#pragma unroll
  for (int j = 0; j < 4; ++j) {
    int n = n0 + j * 16 + lrow;
    float bias = b_ih[n];
#pragma unroll
    for (int r = 0; r < 8; ++r) {
      int m = m0 + r + ha * 8;                  // m = b*T + t
      int bidx = m >> 10, t = m & (T_ - 1);
      gx[((size_t)t * B_ + bidx) * H3_ + n] = acc[j][r] + bias;  // (T,B,3H): per-step contiguous
    }
  }
}

// ---------- persistent GRU recurrence ---------------------------------------
// 32 WGs x 128 threads (4 waves). WG k owns h columns [16k, 16k+16).
// W_hh slice staged into LDS with async copies (ASYNCcnt); per step: WMMA gh,
// f32 gates, h in registers, fused value head, one grid barrier per step.

__global__ void __launch_bounds__(128) k_gru(const float* __restrict__ gx,
                                             const unsigned short* __restrict__ Whh_bf,
                                             const float* __restrict__ b_hh,
                                             const float* __restrict__ W_val,
                                             unsigned short* __restrict__ h_bf,  // 2 buffers of B_*H_
                                             unsigned int* __restrict__ counter,
                                             float* __restrict__ out) {
  __shared__ __attribute__((aligned(16))) unsigned short wsh[3 * 16 * H_];  // 48 KB of 320 KB LDS
  const int wg = blockIdx.x, c0 = wg * 16;
  const int wave = threadIdx.x >> 5, lane = threadIdx.x & 31;
  const int lrow = lane & 15, ha = lane >> 4;
  const int m0 = wave * 16;                      // batch-row tile per wave (4 waves cover B=64)

  // Async DMA stage of this WG's W_hh slice (3 gates x 16 cols x 512 K, bf16) into LDS.
  // For a generic pointer to a __shared__ object, addr[31:0] is the wave-relative
  // LDS byte offset (flat LDS aperture), which is what the async op's VDST expects.
  {
    unsigned lds0 = (unsigned)(size_t)(void*)wsh;
    for (int idx = threadIdx.x; idx < 3 * 16 * (H_ / 8); idx += 128) {  // 3072 x 16B chunks
      int row = idx >> 6, q = idx & 63;
      int g = row >> 4, n = row & 15;
      const unsigned short* src = Whh_bf + (size_t)(g * H_ + c0 + n) * H_ + q * 8;
      unsigned dst = lds0 + (unsigned)idx * 16u;
      asm volatile("global_load_async_to_lds_b128 %0, %1, off"
                   :: "v"(dst), "v"((unsigned long long)(size_t)src)
                   : "memory");
    }
    asm volatile("s_wait_asynccnt 0x0" ::: "memory");
  }
  __syncthreads();

  const int gcol = c0 + lrow;                    // owned h column for this lane
  const float bh_r = b_hh[gcol], bh_z = b_hh[H_ + gcol], bh_n = b_hh[2 * H_ + gcol];
  const float wv = W_val[gcol];

  v8f hreg = {};                                 // f32 master copy of owned h tile
  for (int t = 0; t < T_; ++t) {
    const unsigned short* h_rd = h_bf + (size_t)(t & 1) * (B_ * H_);        // h_{t-1}
    unsigned short*       h_wr = h_bf + (size_t)((t + 1) & 1) * (B_ * H_);  // h_t (double buffer)
    const float* gx_t = gx + (size_t)t * (B_ * H3_);

    v8f accr = {}, accz = {}, accn = {};
    v16bf a = load_a16x32(h_rd, m0, H_, 0, lrow, ha);
#pragma unroll
    for (int kk = 0; kk < 15; ++kk) {            // pipelined: next A (global) before WMMAs
      v16bf br = load_b32x16(wsh + 0 * 16 * H_, 0, H_, kk * 32, lrow, ha);
      v16bf bz = load_b32x16(wsh + 1 * 16 * H_, 0, H_, kk * 32, lrow, ha);
      v16bf bn = load_b32x16(wsh + 2 * 16 * H_, 0, H_, kk * 32, lrow, ha);
      v16bf a2 = load_a16x32(h_rd, m0, H_, (kk + 1) * 32, lrow, ha);
      accr = WMMA_BF16(a, br, accr);
      accz = WMMA_BF16(a, bz, accz);
      accn = WMMA_BF16(a, bn, accn);
      a = a2;
    }
    {
      v16bf br = load_b32x16(wsh + 0 * 16 * H_, 0, H_, 15 * 32, lrow, ha);
      v16bf bz = load_b32x16(wsh + 1 * 16 * H_, 0, H_, 15 * 32, lrow, ha);
      v16bf bn = load_b32x16(wsh + 2 * 16 * H_, 0, H_, 15 * 32, lrow, ha);
      accr = WMMA_BF16(a, br, accr);
      accz = WMMA_BF16(a, bz, accz);
      accn = WMMA_BF16(a, bn, accn);
    }

#pragma unroll
    for (int r = 0; r < 8; ++r) {
      int bidx = m0 + r + ha * 8;                // batch index of this D element
      const float* gp = gx_t + (size_t)bidx * H3_ + gcol;
      float xr = gp[0], xz = gp[H_], xn = gp[2 * H_];
      float rg = 1.f / (1.f + __expf(-(xr + accr[r] + bh_r)));
      float zg = 1.f / (1.f + __expf(-(xz + accz[r] + bh_z)));
      float pre = xn + rg * (accn[r] + bh_n);
      float e2 = __expf(2.f * pre);
      float ng = 1.f - 2.f / (e2 + 1.f);         // tanh, saturates correctly at +/-1
      float hv = (1.f - zg) * ng + zg * hreg[r];
      hreg[r] = hv;
      h_wr[(size_t)bidx * H_ + gcol] = f2bf(hv); // publish bf16 h_t for all WGs
      // fused value head: reduce hv*W_val over the 16 columns held by this lane group
      float pv = hv * wv;
#pragma unroll
      for (int s = 1; s < 16; s <<= 1) pv += __shfl_xor(pv, s, 32);
      if (lrow == 0) atomicAdd(&out[(size_t)bidx * T_ + t], pv);
    }

    __syncthreads();
    if (t + 1 < T_) {
      if (threadIdx.x == 0) {                    // one release/acquire barrier per step
        unsigned int target = (unsigned int)(t + 1) * NWG_GRU;
        __hip_atomic_fetch_add(counter, 1u, __ATOMIC_RELEASE, __HIP_MEMORY_SCOPE_AGENT);
        while (__hip_atomic_load(counter, __ATOMIC_ACQUIRE, __HIP_MEMORY_SCOPE_AGENT) < target)
          __builtin_amdgcn_s_sleep(2);
      }
      __syncthreads();
      __threadfence();                           // make peers' h_t stores visible to vector loads
      // warm L2/L0 for next step's gx slice while waves ramp the GEMM
      __builtin_prefetch(gx + (size_t)(t + 1) * (B_ * H3_) + (size_t)(m0 + lrow) * H3_ + gcol, 0, 1);
    }
  }
}

// ---------- host launcher ----------------------------------------------------

extern "C" void kernel_launch(void* const* d_in, const int* in_sizes, int n_in,
                              void* d_out, int out_size, void* d_ws, size_t ws_size,
                              hipStream_t stream) {
  (void)in_sizes; (void)n_in; (void)out_size; (void)ws_size;
  const float* obs   = (const float*)d_in[0];
  const float* W_emb = (const float*)d_in[1];
  const float* b_emb = (const float*)d_in[2];
  const float* W_ih  = (const float*)d_in[3];
  const float* W_hh  = (const float*)d_in[4];
  const float* b_ih  = (const float*)d_in[5];
  const float* b_hh  = (const float*)d_in[6];
  const float* W_val = (const float*)d_in[7];
  const float* b_val = (const float*)d_in[8];
  float* out = (float*)d_out;

  char* ws = (char*)d_ws;
  size_t off = 0;
  auto take = [&](size_t bytes) -> void* {
    void* p = ws + off;
    off = (off + bytes + 255) & ~(size_t)255;
    return p;
  };
  unsigned short* obs_bf  = (unsigned short*)take((size_t)M_ * OBS_ * 2);   //  16.8 MB
  unsigned short* Wemb_bf = (unsigned short*)take((size_t)H_ * OBS_ * 2);   //   0.1 MB
  unsigned short* Wih_bf  = (unsigned short*)take((size_t)H3_ * H_ * 2);    //   1.5 MB
  unsigned short* Whh_bf  = (unsigned short*)take((size_t)H3_ * H_ * 2);    //   1.5 MB
  unsigned short* x_bf    = (unsigned short*)take((size_t)M_ * H_ * 2);     //  67.1 MB
  float*          gx      = (float*)take((size_t)M_ * H3_ * 4);             // 402.7 MB
  unsigned short* h_bf    = (unsigned short*)take((size_t)2 * B_ * H_ * 2); //   0.1 MB
  unsigned int*   counter = (unsigned int*)take(256);

  k_cvt<<<(M_ * OBS_ + 255) / 256, 256, 0, stream>>>(obs, obs_bf, M_ * OBS_);
  k_cvt<<<(H_ * OBS_ + 255) / 256, 256, 0, stream>>>(W_emb, Wemb_bf, H_ * OBS_);
  k_cvt<<<(H3_ * H_ + 255) / 256, 256, 0, stream>>>(W_ih, Wih_bf, H3_ * H_);
  k_cvt<<<(H3_ * H_ + 255) / 256, 256, 0, stream>>>(W_hh, Whh_bf, H3_ * H_);
  k_init<<<M_ / 256, 256, 0, stream>>>(out, b_val, (unsigned int*)h_bf, counter);

  k_embed<<<(M_ / 16) * 8 / 4, 128, 0, stream>>>(obs_bf, Wemb_bf, b_emb, x_bf);
  k_gx<<<(M_ / 16) * 24 / 4, 128, 0, stream>>>(x_bf, Wih_bf, b_ih, gx);
  k_gru<<<NWG_GRU, 128, 0, stream>>>(gx, Whh_bf, b_hh, W_val, h_bf, counter, out);
}
